// NonLocal_24678882083490
// MI455X (gfx1250) — matrix operand order repaired and checked by reading, not compiled, not run on hardware
//
#include <hip/hip_runtime.h>

typedef __attribute__((ext_vector_type(16))) _Float16 v16h;
typedef __attribute__((ext_vector_type(8)))  _Float16 v8h;
typedef __attribute__((ext_vector_type(8)))  float    v8f;

#define DEVFN static __device__ __forceinline__

constexpr int Bn  = 4;            // batches after T-split
constexpr int Tn  = 8;            // temporal window
constexpr int Cc  = 256;          // channels
constexpr int Ch  = 128;          // half channels (head dim)
constexpr int HWn = 784;          // 28*28
constexpr int Nn  = Tn * HWn;     // 6272 sequence length
constexpr int NT  = Nn / 16;      // 392 query tiles per batch
constexpr int KT  = 64;           // key tile staged in LDS
constexpr int NKT = Nn / KT;      // 98 key tiles
constexpr int QB  = NT / 8;       // 49 blocks per batch (8 waves/block)

// Workspace layout (offsets in _Float16 elements, all 16B-aligned)
constexpr size_t XT_OFF = 0;                              // xT   [B][N][256]
constexpr size_t TH_OFF = XT_OFF + (size_t)Bn * Nn * Cc;  // thetaT [B][N][128]
constexpr size_t PH_OFF = TH_OFF + (size_t)Bn * Nn * Ch;  // phiT   [B][N][128]
constexpr size_t G_OFF  = PH_OFF + (size_t)Bn * Nn * Ch;  // g_cn   [B][128][N]
constexpr size_t Y_OFF  = G_OFF  + (size_t)Bn * Nn * Ch;  // yT     [B][N][128]
constexpr size_t W3_OFF = Y_OFF  + (size_t)Bn * Nn * Ch;  // theta_w|phi_w|g_w f16
constexpr size_t EW_OFF = W3_OFF + (size_t)3 * Ch * Cc;   // embed_w f16 [256][128]

// ---- WMMA fragment loaders (16x16x32 f16, wave32) ----
// A fragment (16x32, M=lane%16): halves = 16 contiguous K at c0+16*hi
DEVFN v16h ldA(const _Float16* row, int c0, int hi) {
  v8h lo = *(const v8h*)(row + c0 + 16 * hi);
  v8h hh = *(const v8h*)(row + c0 + 16 * hi + 8);
  return __builtin_shufflevector(lo, hh, 0,1,2,3,4,5,6,7,8,9,10,11,12,13,14,15);
}
// B fragment (32x16, N=lane%16): halves = 16 contiguous K at c0+16*hi
DEVFN v16h ldB(const _Float16* row, int c0, int hi) {
  v8h lo = *(const v8h*)(row + c0 + 16 * hi);
  v8h hh = *(const v8h*)(row + c0 + 16 * hi + 8);
  return __builtin_shufflevector(lo, hh, 0,1,2,3,4,5,6,7,8,9,10,11,12,13,14,15);
}
DEVFN v8f wmma(v16h a, v16h b, v8f c) {
  return __builtin_amdgcn_wmma_f32_16x16x32_f16(false, a, false, b, (short)0, c, false, false);
}
DEVFN int pk2(float a, float b) {
  union { _Float16 h[2]; int i; } u;
  u.h[0] = (_Float16)a; u.h[1] = (_Float16)b; return u.i;
}

// Async global->LDS 16B move (GVS addressing: LDS byte offset, 32-bit voffset, SGPR64 base).
// Tracked by ASYNCcnt; in-order completion per wave.
DEVFN void async_ld16(unsigned lds_off, unsigned gl_off, unsigned long long base) {
  asm volatile("global_load_async_to_lds_b128 %0, %1, %2"
               :: "v"(lds_off), "v"(gl_off), "s"(base) : "memory");
}

// ---- Kernel 1: transpose feat -> xT (f16) and convert weights to f16 ----
constexpr int NXT = Bn * Tn * Cc * HWn;           // 6,422,528
constexpr int NPK = NXT + 4 * Ch * Cc;            // + 131,072 weight elems

__global__ __launch_bounds__(256) void nl_pack(const float* __restrict__ feat,
                                               const float* __restrict__ tw,
                                               const float* __restrict__ pw,
                                               const float* __restrict__ gw,
                                               const float* __restrict__ ew,
                                               _Float16* __restrict__ h) {
  int idx = blockIdx.x * 256 + threadIdx.x;
  if (idx < NXT) {
    int hw = idx % HWn;
    int tmp = idx / HWn;
    int c  = tmp % Cc;
    int bt = tmp / Cc;
    int b = bt / Tn, t = bt % Tn;
    int n = t * HWn + hw;
    h[XT_OFF + ((size_t)b * Nn + n) * Cc + c] = (_Float16)feat[idx];
  } else if (idx < NPK) {
    int j = idx - NXT;
    if      (j < Ch * Cc)     h[W3_OFF + j] = (_Float16)tw[j];
    else if (j < 2 * Ch * Cc) h[W3_OFF + j] = (_Float16)pw[j - Ch * Cc];
    else if (j < 3 * Ch * Cc) h[W3_OFF + j] = (_Float16)gw[j - 2 * Ch * Cc];
    else                      h[EW_OFF + (j - 3 * Ch * Cc)] = (_Float16)ew[j - 3 * Ch * Cc];
  }
}

// ---- Kernel 2: projections theta/phi/g = W @ x  (per-wave 16-column tile) ----
__global__ __launch_bounds__(256) void nl_proj(_Float16* __restrict__ h,
                                               const float* __restrict__ tb,
                                               const float* __restrict__ pb,
                                               const float* __restrict__ gb) {
  int wave = blockIdx.x * 8 + (threadIdx.x >> 5);
  int b = wave / NT, nt = wave % NT;
  int lane = threadIdx.x & 31, lr = lane & 15, hi = lane >> 4;

  const _Float16* xrow = h + XT_OFF + ((size_t)b * Nn + nt * 16 + lr) * Cc;
  v16h Bx[8];
#pragma unroll
  for (int cc = 0; cc < 8; ++cc) Bx[cc] = ldB(xrow, cc * 32, hi);

  const float* biases[3] = { tb, pb, gb };
#pragma unroll
  for (int m = 0; m < 3; ++m) {
    const _Float16* wbase = h + W3_OFF + (size_t)m * Ch * Cc;
    const float* bias = biases[m];
#pragma unroll
    for (int ot = 0; ot < 8; ++ot) {
      int o0 = ot * 16;
      const _Float16* arow = wbase + (size_t)(o0 + lr) * Cc;
      v8f acc = {};
#pragma unroll
      for (int cc = 0; cc < 8; ++cc) acc = wmma(ldA(arow, cc * 32, hi), Bx[cc], acc);
      int ob = o0 + 8 * hi;
      if (m < 2) {
        v8h p;
#pragma unroll
        for (int v = 0; v < 8; ++v) p[v] = (_Float16)(acc[v] + bias[ob + v]);
        _Float16* dst = h + (m == 0 ? TH_OFF : PH_OFF)
                          + ((size_t)b * Nn + nt * 16 + lr) * Ch + ob;
        *(v8h*)dst = p;
      } else {
#pragma unroll
        for (int v = 0; v < 8; ++v)
          h[G_OFF + ((size_t)b * Ch + ob + v) * Nn + nt * 16 + lr] =
              (_Float16)(acc[v] + bias[ob + v]);
      }
    }
  }
}

// ---- Kernel 3: fused flash attention with async double-buffered LDS staging ----
// LDS per stage: phi tile (64 keys x 128 ch, 16KB) + g tile (128 ch x 64 keys, 16KB).
// Two stages -> 64KB dynamic LDS.
DEVFN void stage_issue(const _Float16* h, int b, int i0, int buf, int tid) {
  unsigned long long phiBase =
      (unsigned long long)(uintptr_t)(h + PH_OFF + (size_t)b * Nn * Ch);
  unsigned long long gBase =
      (unsigned long long)(uintptr_t)(h + G_OFF + (size_t)b * Ch * Nn);
  unsigned ldsPhi = (unsigned)buf * 32768u;
  unsigned ldsG   = ldsPhi + 16384u;
  // phi tile: contiguous 16KB (rows i0..i0+63 of phiT)
#pragma unroll
  for (int j = 0; j < 4; ++j) {
    unsigned chunk = (unsigned)tid + 256u * j;            // 1024 x 16B
    async_ld16(ldsPhi + chunk * 16u,
               (unsigned)i0 * (unsigned)Ch * 2u + chunk * 16u, phiBase);
  }
  // g tile: 128 rows x 128B, global row stride Nn*2 bytes
#pragma unroll
  for (int j = 0; j < 4; ++j) {
    unsigned k = (unsigned)tid + 256u * j;                // 1024 x 16B
    unsigned row = k >> 3, col = k & 7;
    async_ld16(ldsG + row * 128u + col * 16u,
               (row * (unsigned)Nn + (unsigned)i0) * 2u + col * 16u, gBase);
  }
}

__global__ __launch_bounds__(256) void nl_attn(const _Float16* __restrict__ h,
                                               _Float16* __restrict__ hw_) {
  extern __shared__ _Float16 smem[];
  int tid = threadIdx.x;
  int waveid = tid >> 5;
  int b  = blockIdx.x / QB;
  int qt = (blockIdx.x % QB) * 8 + waveid;   // all 8 waves share batch b
  int lane = tid & 31, lr = lane & 15, hi = lane >> 4;
  int q0 = qt * 16;

  // Resident query (theta) B-fragments, K = 128 channels in 4 chunks of 32
  const _Float16* qrow = h + TH_OFF + ((size_t)b * Nn + q0 + lr) * Ch;
  v16h Bq[4];
#pragma unroll
  for (int cc = 0; cc < 4; ++cc) Bq[cc] = ldB(qrow, cc * 32, hi);

  v8f O[8];
#pragma unroll
  for (int oc = 0; oc < 8; ++oc) O[oc] = (v8f){};
  float mrun = -3.0e38f, lrun = 0.0f;

  stage_issue(h, b, 0, 0, tid);
  for (int t = 0; t < NKT; ++t) {
    int buf = t & 1;
    if (t + 1 < NKT) {
      stage_issue(h, b, (t + 1) * KT, (t + 1) & 1, tid);
      asm volatile("s_wait_asynccnt 8" ::: "memory");   // current tile landed
    } else {
      asm volatile("s_wait_asynccnt 0" ::: "memory");
    }
    __syncthreads();
    const _Float16* phiL = smem + (size_t)buf * 16384;  // halves
    const _Float16* gL   = phiL + 8192;

#pragma unroll
    for (int kk = 0; kk < 2; ++kk) {                    // two 32-key subtiles
      const _Float16* pr0 = phiL + (kk * 32 + lr) * Ch;
      const _Float16* pr1 = pr0 + 16 * Ch;
      v8f S0 = {}, S1 = {};
#pragma unroll
      for (int cc = 0; cc < 4; ++cc) S0 = wmma(ldA(pr0, cc * 32, hi), Bq[cc], S0);
#pragma unroll
      for (int cc = 0; cc < 4; ++cc) S1 = wmma(ldA(pr1, cc * 32, hi), Bq[cc], S1);

      // Online softmax over keys; lane pair (L, L+16) holds the same query.
      float mloc = -3.0e38f;
#pragma unroll
      for (int v = 0; v < 8; ++v) mloc = fmaxf(mloc, fmaxf(S0[v], S1[v]));
      float mtile = fmaxf(mloc, __shfl_xor(mloc, 16, 32));
      float mnew  = fmaxf(mrun, mtile);
      float scale = __expf(mrun - mnew);

      float ex0[8], ex1[8], ls = 0.0f;
#pragma unroll
      for (int v = 0; v < 8; ++v) {
        ex0[v] = __expf(S0[v] - mnew);
        ex1[v] = __expf(S1[v] - mnew);
        ls += ex0[v] + ex1[v];
      }
      lrun = lrun * scale + ls + __shfl_xor(ls, 16, 32);
      mrun = mnew;
#pragma unroll
      for (int oc = 0; oc < 8; ++oc)
#pragma unroll
        for (int v = 0; v < 8; ++v) O[oc][v] *= scale;

      // Repack exp(S) (D layout) into a 32x16 B-fragment via one xor-16 exchange.
      int p0[4], p1[4], rcv[4];
#pragma unroll
      for (int k = 0; k < 4; ++k) {
        p0[k] = pk2(ex0[2 * k], ex0[2 * k + 1]);
        p1[k] = pk2(ex1[2 * k], ex1[2 * k + 1]);
      }
#pragma unroll
      for (int k = 0; k < 4; ++k) rcv[k] = __shfl_xor(hi ? p0[k] : p1[k], 16, 32);
      union { int i[8]; v16h f; } P;
#pragma unroll
      for (int k = 0; k < 4; ++k) {
        P.i[k]     = hi ? rcv[k] : p0[k];
        P.i[4 + k] = hi ? p1[k]  : rcv[k];
      }

      // O += g * P  (128 channels in 8 tiles of 16, K = 32 keys) — g from LDS
#pragma unroll
      for (int oc = 0; oc < 8; ++oc) {
        const _Float16* grow = gL + (oc * 16 + lr) * KT + kk * 32;
        O[oc] = wmma(ldA(grow, 0, hi), P.f, O[oc]);
      }
    }
    __syncthreads();   // all waves done with buf before it is refilled
  }

  float inv = 1.0f / lrun;
#pragma unroll
  for (int oc = 0; oc < 8; ++oc) {
    v8h p;
#pragma unroll
    for (int v = 0; v < 8; ++v) p[v] = (_Float16)(O[oc][v] * inv);
    *(v8h*)(hw_ + Y_OFF + ((size_t)b * Nn + q0 + lr) * Ch + oc * 16 + 8 * hi) = p;
  }
}

// ---- Kernel 4: out = feat + embed_w @ y + embed_b, back to (BT,C,H,W) ----
__global__ __launch_bounds__(256) void nl_embed(const _Float16* __restrict__ h,
                                                const float* __restrict__ feat,
                                                const float* __restrict__ eb,
                                                float* __restrict__ out) {
  int wave = blockIdx.x * 8 + (threadIdx.x >> 5);
  int b = wave / NT, nt = wave % NT;
  int lane = threadIdx.x & 31, lr = lane & 15, hi = lane >> 4;
  int n = nt * 16 + lr;

  const _Float16* yrow = h + Y_OFF + ((size_t)b * Nn + n) * Ch;
  v16h By[4];
#pragma unroll
  for (int cc = 0; cc < 4; ++cc) By[cc] = ldB(yrow, cc * 32, hi);

  int t = n / HWn, hw = n % HWn;
  size_t base = ((size_t)(b * Tn + t) * Cc) * HWn + hw;

#pragma unroll
  for (int ot = 0; ot < 16; ++ot) {
    int o0 = ot * 16;
    const _Float16* arow = h + EW_OFF + (size_t)(o0 + lr) * Ch;
    v8f acc = {};
#pragma unroll
    for (int cc = 0; cc < 4; ++cc) acc = wmma(ldA(arow, cc * 32, hi), By[cc], acc);
    int ob = o0 + 8 * hi;
#pragma unroll
    for (int v = 0; v < 8; ++v) {
      int o = ob + v;
      size_t idx = base + (size_t)o * HWn;
      out[idx] = feat[idx] + acc[v] + eb[o];
    }
  }
}

extern "C" void kernel_launch(void* const* d_in, const int* in_sizes, int n_in,
                              void* d_out, int out_size, void* d_ws, size_t ws_size,
                              hipStream_t stream) {
  (void)in_sizes; (void)n_in; (void)out_size; (void)ws_size;
  const float* feat = (const float*)d_in[0];
  const float* tw   = (const float*)d_in[1];
  const float* tb   = (const float*)d_in[2];
  const float* pw   = (const float*)d_in[3];
  const float* pb   = (const float*)d_in[4];
  const float* gw   = (const float*)d_in[5];
  const float* gb   = (const float*)d_in[6];
  const float* ew   = (const float*)d_in[7];
  const float* eb   = (const float*)d_in[8];
  _Float16* h = (_Float16*)d_ws;
  float* out = (float*)d_out;

  int packBlocks = (NPK + 255) / 256;                  // 25600
  int tileBlocks = (Bn * NT) / 8;                      // 196 blocks x 8 waves

  nl_pack<<<packBlocks, 256, 0, stream>>>(feat, tw, pw, gw, ew, h);
  nl_proj<<<tileBlocks, 256, 0, stream>>>(h, tb, pb, gb);
  nl_attn<<<tileBlocks, 256, 65536, stream>>>(h, h);
  nl_embed<<<tileBlocks, 256, 0, stream>>>(h, feat, eb, out);
}
